// SNN_MLP_Rate_54004918780360
// MI455X (gfx1250) — matrix operand order repaired and checked by reading, not compiled, not run on hardware
//
#include <hip/hip_runtime.h>
#include <hip/hip_bf16.h>

// ---------------------------------------------------------------------------
// SNN MLP (rate-coded) fused over T=50 timesteps, f16 WMMA on gfx1250.
//   layer1: [16x784] x [784x128] per block-tile, K padded to 800 (25 x k32)
//   layer2: [16x128] x [128x16(pad of 10)]       (4 x k32), wave 0 only
// mem1/mem2 LIF state lives in v8f WMMA accumulator fragments across all t.
// ---------------------------------------------------------------------------

typedef __attribute__((ext_vector_type(16))) _Float16 v16h;
typedef __attribute__((ext_vector_type(8)))  _Float16 v8h;
typedef __attribute__((ext_vector_type(4)))  _Float16 v4h;
typedef __attribute__((ext_vector_type(8)))  float    v8f;

#define T_STEPS 50
#define BATCH   1024
#define IN_F    784
#define HID     128
#define OUTP    10
#define NKT1    25      // k-chunks of 32 for layer 1 (784 padded to 800)
#define NKT2    4       // k-chunks of 32 for layer 2 (K = 128)
#define NT1     8       // hidden-column tiles (128 / 16)
#define BT      16      // batch rows per workgroup
#define XROW    800     // halves per padded x row in LDS
#define SROW    136     // halves per spk1 row in LDS (128 + pad, 272B = 17*16)

#define FRAG_W1_HALVES (NT1 * NKT1 * 32 * 16)   // 102400 (200 KB in ws)
#define FRAG_W2_HALVES (NKT2 * 32 * 16)         // 2048   (4 KB in ws)

// Build a v16h A/B fragment register from two contiguous 16B LDS chunks:
// each lane needs halves [p..p+15] (contiguous K order, matching packed B).
__device__ __forceinline__ v16h ldsA(const _Float16* p) {
  v8h lo = *(const v8h*)(p);
  v8h hi = *(const v8h*)(p + 8);
  return __builtin_shufflevector(lo, hi, 0,1,2,3,4,5,6,7,8,9,10,11,12,13,14,15);
}

__device__ __forceinline__ v8f splat8(float v) {
  v8f r;
#pragma unroll
  for (int i = 0; i < 8; ++i) r[i] = v;
  return r;
}

// ---------------------------------------------------------------------------
// Pre-pack W1 [128,784] f32 and W2 [10,128] f32 into f16 fragment-major
// buffers so the main kernel's B-matrix loads are coalesced b128 loads in
// the contiguous-K B lane layout this kernel uses (col = lane&15, K half-run chosen by lane>=16).
// ---------------------------------------------------------------------------
__global__ __launch_bounds__(256) void snn_pack_weights(
    const float* __restrict__ W1, const float* __restrict__ W2,
    _Float16* __restrict__ fw1, _Float16* __restrict__ fw2)
{
  int id = blockIdx.x * 256 + threadIdx.x;
  if (id < NT1 * NKT1 * 32) {
    int lane = id & 31;
    int g    = id >> 5;
    int kt   = g % NKT1;
    int nt   = g / NKT1;
    int n    = nt * 16 + (lane & 15);
    int k0   = kt * 32 + ((lane & 16) ? 16 : 0);
#pragma unroll
    for (int j = 0; j < 16; ++j) {
      int k = k0 + j;
      fw1[(size_t)id * 16 + j] = (k < IN_F) ? (_Float16)W1[n * IN_F + k]
                                            : (_Float16)0.0f;
    }
  } else if (id < NT1 * NKT1 * 32 + NKT2 * 32) {
    int id2  = id - NT1 * NKT1 * 32;
    int lane = id2 & 31;
    int kt   = id2 >> 5;
    int row  = lane & 15;                      // output column (pad 10..15 = 0)
    int k0   = kt * 32 + ((lane & 16) ? 16 : 0);
#pragma unroll
    for (int j = 0; j < 16; ++j) {
      fw2[id2 * 16 + j] = (row < OUTP) ? (_Float16)W2[row * HID + k0 + j]
                                       : (_Float16)0.0f;
    }
  }
}

// ---------------------------------------------------------------------------
// Main fused SNN kernel. One block = 16 batch rows, 8 waves. Wave w computes
// hidden columns [16w, 16w+16). Wave 0 additionally runs layer 2.
// ---------------------------------------------------------------------------
__global__ __launch_bounds__(256) void snn_main(
    const float* __restrict__ x,    // [T, B, 784] binary spikes, f32
    const float* __restrict__ b1,   // [128]
    const float* __restrict__ b2,   // [10]
    const _Float16* __restrict__ fw1,
    const _Float16* __restrict__ fw2,
    float* __restrict__ out)        // [T, B, 10]
{
  __shared__ __align__(16) _Float16 Xh[BT * XROW];  // 25.6 KB spike tile (f16)
  __shared__ __align__(16) _Float16 Sh[BT * SROW];  // 4.3 KB  spk1 tile (f16)

  const int tid   = threadIdx.x;
  const int w     = tid >> 5;           // wave id 0..7 (uniform per wave)
  const int lane  = tid & 31;
  const int col   = lane & 15;          // N within 16x16 tile
  const int rbase = (lane & 16) ? 8 : 0;// C/D fragment row offset per lane half
  const int kofs  = (lane & 16) ? 16 : 0;// A fragment K offset per lane half
  const int bbase = blockIdx.x * BT;

  const float b1v = b1[w * 16 + col];
  const float b2v = (col < OUTP) ? b2[col] : 0.0f;

  v8f mem1 = splat8(0.0f);              // persistent LIF state, layer 1
  v8f mem2 = splat8(0.0f);              // persistent LIF state, layer 2 (wave 0)

  // x-tile loader assignment: 16 threads per batch row, 196 float4 per row
  const int lrow = tid >> 4;            // 0..15
  const int lj   = tid & 15;            // 0..15

  const _Float16* aBase = Xh + col * XROW + kofs;   // layer-1 A fragment base
  const _Float16* sBase = Sh + col * SROW + kofs;   // layer-2 A fragment base
  const v16h* w1p = (const v16h*)fw1 + (size_t)(w * NKT1) * 32 + lane;

#pragma unroll 1
  for (int t = 0; t < T_STEPS; ++t) {
    __syncthreads();  // prior iteration's consumers of Xh/Sh are done

    // ---- stage spike tile: f32 global -> f16 LDS (pad K 784..799 = 0) ----
    {
      const float* xrow = x + ((size_t)t * BATCH + bbase + lrow) * IN_F;
      _Float16* dst = Xh + lrow * XROW;
#pragma unroll
      for (int it = 0; it < 13; ++it) {
        int jj = lj + it * 16;
        if (jj < 196) {
          float4 v = *(const float4*)(xrow + jj * 4);
          v4h p = { (_Float16)v.x, (_Float16)v.y, (_Float16)v.z, (_Float16)v.w };
          *(v4h*)(dst + jj * 4) = p;
        }
      }
      if (lj < 2) {
        uint4 z = {0u, 0u, 0u, 0u};
        *(uint4*)(dst + 784 + lj * 8) = z;
      }
    }

    // prefetch next timestep's tile into cache while we compute this one
    if (t + 1 < T_STEPS) {
      const float* nx = x + ((size_t)(t + 1) * BATCH + bbase) * IN_F;
#pragma unroll
      for (int it = 0; it < 2; ++it) {
        int j = tid + it * 256;                 // 392 x 128B lines = 50 KB
        if (j < 392) __builtin_prefetch(nx + (size_t)j * 32, 0, 0);
      }
    }

    __syncthreads();  // Xh ready

    // ---- layer 1: cur1 = x @ W1^T + b1  (25 x wmma_f32_16x16x32_f16) ----
    v8f acc = splat8(b1v);
#pragma unroll
    for (int kt = 0; kt < NKT1; ++kt) {
      v16h bf = w1p[kt * 32];
      v16h af = ldsA(aBase + kt * 32);
      acc = __builtin_amdgcn_wmma_f32_16x16x32_f16(
          false, af, false, bf, (short)0, acc, false, false);
    }

    // ---- LIF 1 (reset-by-subtract with previous mem) + spk1 -> LDS ----
    {
      int sb = rbase * SROW + w * 16 + col;
#pragma unroll
      for (int i = 0; i < 8; ++i) {
        float rst = (mem1[i] > 1.0f) ? 1.0f : 0.0f;
        float m   = 0.9f * mem1[i] + acc[i] - rst;
        mem1[i]   = m;
        Sh[sb + i * SROW] = (_Float16)((m > 1.0f) ? 1.0f : 0.0f);
      }
    }

    __syncthreads();  // Sh ready

    // ---- layer 2 (wave 0): cur2 = spk1 @ W2^T + b2, LIF 2, emit spk2 ----
    if (w == 0) {
      v8f acc2 = splat8(b2v);
#pragma unroll
      for (int kt = 0; kt < NKT2; ++kt) {
        v16h af = ldsA(sBase + kt * 32);
        v16h bf = *(const v16h*)(fw2 + (kt * 32 + lane) * 16);
        acc2 = __builtin_amdgcn_wmma_f32_16x16x32_f16(
            false, af, false, bf, (short)0, acc2, false, false);
      }
      size_t ob = ((size_t)t * BATCH + bbase + rbase) * OUTP + col;
#pragma unroll
      for (int i = 0; i < 8; ++i) {
        float rst = (mem2[i] > 1.0f) ? 1.0f : 0.0f;
        float m   = 0.9f * mem2[i] + acc2[i] - rst;
        mem2[i]   = m;
        if (col < OUTP)
          out[ob + (size_t)i * OUTP] = (m > 1.0f) ? 1.0f : 0.0f;
      }
    }
  }
}

extern "C" void kernel_launch(void* const* d_in, const int* in_sizes, int n_in,
                              void* d_out, int out_size, void* d_ws, size_t ws_size,
                              hipStream_t stream) {
  (void)in_sizes; (void)n_in; (void)out_size; (void)ws_size;
  const float* x  = (const float*)d_in[0];   // [50,1024,1,28,28]
  const float* W1 = (const float*)d_in[1];   // [128,784]
  const float* b1 = (const float*)d_in[2];   // [128]
  const float* W2 = (const float*)d_in[3];   // [10,128]
  const float* b2 = (const float*)d_in[4];   // [10]
  float* out = (float*)d_out;                // [50,1024,10]

  _Float16* fw1 = (_Float16*)d_ws;
  _Float16* fw2 = fw1 + FRAG_W1_HALVES;      // 204800 B offset (32B aligned)

  int packThreads = NT1 * NKT1 * 32 + NKT2 * 32;   // 6528
  snn_pack_weights<<<(packThreads + 255) / 256, 256, 0, stream>>>(W1, W2, fw1, fw2);
  snn_main<<<BATCH / BT, 256, 0, stream>>>(x, b1, b2, fw1, fw2, out);
}
